// ContextEBM_50525995270535
// MI455X (gfx1250) — matrix operand order repaired and checked by reading, not compiled, not run on hardware
//
#include <hip/hip_runtime.h>

typedef __attribute__((ext_vector_type(16))) _Float16 v16h;
typedef __attribute__((ext_vector_type(8)))  _Float16 v8h;
typedef __attribute__((ext_vector_type(8)))  float    v8f;
typedef __attribute__((ext_vector_type(4)))  float    v4f;

#define MLP_WIDTH 64
#define STEPS 50
#define WAVES_PER_WG 8
#define NTHREADS (WAVES_PER_WG * 32)
#define TILE 16
#define SAMPLES_PER_WG (WAVES_PER_WG * TILE)

// LDS budget (bytes)
// weights: W1,W2,W1T,W2T f16 (4*8192) + W3 f16 (128) + w0x/w0y/w0c/b0/b1/b2 f32 (6*256)
#define WEIGHT_BYTES (4 * 4096 * 2 + 64 * 2 + 6 * 64 * 4)            /* 34432 */
// per-wave: h0,h1,h2,g f16 [16][64] (4*2048) + x,y,c f32 [16] (192) + pad
#define PER_WAVE_BYTES (4 * 16 * MLP_WIDTH * 2 + 3 * 16 * 4 + 64)    /* 8448  */
#define SMEM_BYTES (WEIGHT_BYTES + WAVES_PER_WG * PER_WAVE_BYTES)    /* 102016 */

union U16h { v16h v; v8h h[2]; };
union U8f  { v8f  v; v4f f[2]; };

// A fragment (16x32 f16) of row-major W[64][64], tile (mbase, kbase).
// 16-bit A layout: lanes 0-15 -> M=lane,   elems 0-7 = K=k0..k0+7, elems 8-15 = K=k0+16..k0+23
//                  lanes 16-31 -> M=lane-16, same with +8 K offset.
__device__ __forceinline__ v16h load_a(const _Float16* W, int mbase, int kbase, int lane) {
    const int m    = mbase + (lane & 15);
    const int koff = kbase + ((lane >> 4) << 3);
    U16h u;
    u.h[0] = *(const v8h*)(W + m * MLP_WIDTH + koff);
    u.h[1] = *(const v8h*)(W + m * MLP_WIDTH + koff + 16);
    return u.v;
}

// B fragment (32x16 f16) from activation buffer act[16][64] (sample-major).
// 16-bit B layout: lanes 0-15: N=lane, elems 0-7 = K=k0..k0+7, elems 8-15 = K=k0+16..k0+23; lanes 16-31: K += 8.
__device__ __forceinline__ v16h load_b(const _Float16* act, int kbase, int lane) {
    const int s = lane & 15;
    const int k = kbase + ((lane >> 4) << 3);
    U16h u;
    u.h[0] = *(const v8h*)(act + s * MLP_WIDTH + k);
    u.h[1] = *(const v8h*)(act + s * MLP_WIDTH + k + 16);
    return u.v;
}

// One 64x64 dense layer on a 16-sample tile.
//  FORWARD: out = relu(W @ in + bias)
// !FORWARD: out = (W @ in) .* (mask > 0)       (backward through transposed weights)
// All fragment loads are issued before any WMMA so the scheduler can use partial
// DScnt waits and pipeline 4 independent accumulator chains through the matrix unit.
template<bool FORWARD>
__device__ __forceinline__ void dense64(const _Float16* W, const float* bias,
                                        const _Float16* in, const _Float16* mask,
                                        _Float16* out, int lane) {
    const int s    = lane & 15;
    const int half = (lane >> 4) << 3;

    const v16h bf0 = load_b(in, 0,  lane);
    const v16h bf1 = load_b(in, 32, lane);

    v16h a0[4], a1[4];
    #pragma unroll
    for (int t = 0; t < 4; ++t) {
        a0[t] = load_a(W, t * 16, 0,  lane);
        a1[t] = load_a(W, t * 16, 32, lane);
    }

    v8f acc[4];
    #pragma unroll
    for (int t = 0; t < 4; ++t) {
        if constexpr (FORWARD) {                 // fold bias into the accumulator
            const int fbase = t * 16 + half;
            U8f cu;
            cu.f[0] = *(const v4f*)(bias + fbase);
            cu.f[1] = *(const v4f*)(bias + fbase + 4);
            acc[t] = cu.v;
        } else {
            acc[t] = {};
        }
    }

    // K = 0..31 pass, 4 independent chains
    #pragma unroll
    for (int t = 0; t < 4; ++t)
        acc[t] = __builtin_amdgcn_wmma_f32_16x16x32_f16(false, a0[t], false, bf0,
                                                        (short)0, acc[t], false, false);
    // K = 32..63 pass (dependent WMMAs are 4 ops apart -> hazard window covered)
    #pragma unroll
    for (int t = 0; t < 4; ++t)
        acc[t] = __builtin_amdgcn_wmma_f32_16x16x32_f16(false, a1[t], false, bf1,
                                                        (short)0, acc[t], false, false);

    #pragma unroll
    for (int t = 0; t < 4; ++t) {
        const int fbase = t * 16 + half;          // 8 consecutive features per lane
        v8h o;
        if constexpr (FORWARD) {
            #pragma unroll
            for (int r = 0; r < 8; ++r)
                o[r] = (_Float16)fmaxf(acc[t][r], 0.0f);
        } else {
            const v8h mv = *(const v8h*)(mask + s * MLP_WIDTH + fbase);
            #pragma unroll
            for (int r = 0; r < 8; ++r)
                o[r] = ((float)mv[r] > 0.0f) ? (_Float16)acc[t][r] : (_Float16)0.0f;
        }
        *(v8h*)(out + s * MLP_WIDTH + fbase) = o; // D layout: lane=sample, VGPRs=8 features
    }
}

__global__ __launch_bounds__(NTHREADS)
void ContextEBM_50525995270535_kernel(const float* __restrict__ xg,
                                      const float* __restrict__ W0g, const float* __restrict__ b0g,
                                      const float* __restrict__ W1g, const float* __restrict__ b1g,
                                      const float* __restrict__ W2g, const float* __restrict__ b2g,
                                      const float* __restrict__ W3g, const float* __restrict__ b3g,
                                      float* __restrict__ outg) {
    extern __shared__ __align__(16) char smem[];
    _Float16* W1l  = (_Float16*)smem;           // [64][64] row-major
    _Float16* W2l  = W1l + 4096;
    _Float16* W1Tl = W2l + 4096;                // W1T[i][o] = W1[o][i]
    _Float16* W2Tl = W1Tl + 4096;
    _Float16* W3l  = W2Tl + 4096;               // [64]
    float* w0x = (float*)(W3l + 64);            // W0 columns + biases, f32 [64] each
    float* w0y = w0x + 64;
    float* w0c = w0y + 64;
    float* b0l = w0c + 64;
    float* b1l = b0l + 64;
    float* b2l = b1l + 64;
    char*  wavemem = (char*)(b2l + 64);

    const int tid  = threadIdx.x;
    const int wave = tid >> 5;
    const int lane = tid & 31;
    const int sampleBase = blockIdx.x * SAMPLES_PER_WG;

    // ---- one-time weight staging: f32 -> f16, plus explicit transposes ----
    for (int i = tid; i < 4096; i += NTHREADS) {
        const int o = i >> 6, k = i & 63;
        const float w1 = W1g[i], w2 = W2g[i];
        W1l[i] = (_Float16)w1;  W2l[i] = (_Float16)w2;
        W1Tl[k * 64 + o] = (_Float16)w1;
        W2Tl[k * 64 + o] = (_Float16)w2;
    }
    if (tid < 64) {
        w0x[tid] = W0g[tid * 3 + 0];
        w0y[tid] = W0g[tid * 3 + 1];
        w0c[tid] = W0g[tid * 3 + 2];
        b0l[tid] = b0g[tid];
        b1l[tid] = b1g[tid];
        b2l[tid] = b2g[tid];
        W3l[tid] = (_Float16)W3g[tid];
    }
    if (tid < SAMPLES_PER_WG) {                 // init per-wave x / y / c state
        const int w = tid >> 4, s = tid & 15;
        float* xs = (float*)(wavemem + w * PER_WAVE_BYTES + 4 * 16 * MLP_WIDTH * 2);
        xs[s]      = xg[sampleBase + tid];
        xs[16 + s] = 0.0f;                      // y0 = Y_MEAN = 0
        xs[32 + s] = 0.0f;                      // c0 = 0
    }
    __syncthreads();

    // ---- per-wave private staging buffers ----
    char* wb = wavemem + wave * PER_WAVE_BYTES;
    _Float16* h0 = (_Float16*)wb;               // [16][64] post-relu acts (sample-major)
    _Float16* h1 = h0 + 16 * MLP_WIDTH;
    _Float16* h2 = h1 + 16 * MLP_WIDTH;
    _Float16* gb = h2 + 16 * MLP_WIDTH;         // gradient staging (reused g2->g1->g0)
    float* Xs = (float*)(gb + 16 * MLP_WIDTH);
    float* Ys = Xs + 16;
    float* Cs = Ys + 16;

    #pragma unroll 1
    for (int step = 0; step < STEPS; ++step) {
        // layer 0 (64x3): h0 = relu(W0*[x,y,c] + b0); lane -> (sample, 32-feature half)
        {
            const int s = lane >> 1, fb = (lane & 1) << 5;
            const float xv = Xs[s], yv = Ys[s], cv = Cs[s];
            #pragma unroll
            for (int j = 0; j < 32; j += 8) {
                v8h o;
                #pragma unroll
                for (int q = 0; q < 2; ++q) {   // 4 features per v4f group
                    const int f = fb + j + q * 4;
                    const v4f cx = *(const v4f*)(w0x + f);
                    const v4f cy = *(const v4f*)(w0y + f);
                    const v4f cc = *(const v4f*)(w0c + f);
                    const v4f cb = *(const v4f*)(b0l + f);
                    #pragma unroll
                    for (int t = 0; t < 4; ++t) {
                        const float pre = cx[t] * xv + cy[t] * yv + cc[t] * cv + cb[t];
                        o[q * 4 + t] = (_Float16)fmaxf(pre, 0.0f);
                    }
                }
                *(v8h*)(h0 + s * MLP_WIDTH + fb + j) = o;
            }
        }
        dense64<true>(W1l, b1l, h0, nullptr, h1, lane);
        dense64<true>(W2l, b2l, h1, nullptr, h2, lane);

        // backward head: g2 = W3 .* 1[h2 > 0]
        {
            const int s = lane >> 1, fb = (lane & 1) << 5;
            #pragma unroll
            for (int j = 0; j < 32; j += 8) {
                const v8h hv = *(const v8h*)(h2 + s * MLP_WIDTH + fb + j);
                const v8h wv = *(const v8h*)(W3l + fb + j);
                v8h g;
                #pragma unroll
                for (int t = 0; t < 8; ++t)
                    g[t] = ((float)hv[t] > 0.0f) ? wv[t] : (_Float16)0.0f;
                *(v8h*)(gb + s * MLP_WIDTH + fb + j) = g;
            }
        }
        dense64<false>(W2Tl, nullptr, gb, h1, gb, lane);   // g1 = (W2^T g2) .* 1[h1>0]
        dense64<false>(W1Tl, nullptr, gb, h0, gb, lane);   // g0 = (W1^T g1) .* 1[h0>0]

        // input grads: gy[s] = w0y . g0[s], gc[s] = w0c . g0[s]; SGD update
        {
            const int s = lane & 15;
            const float* wcol = (lane < 16) ? w0y : w0c;
            float acc = 0.0f;
            #pragma unroll
            for (int j = 0; j < MLP_WIDTH; j += 8) {
                const v8h g  = *(const v8h*)(gb + s * MLP_WIDTH + j);
                const v4f wl = *(const v4f*)(wcol + j);
                const v4f wh = *(const v4f*)(wcol + j + 4);
                acc += wl[0] * (float)g[0] + wl[1] * (float)g[1]
                     + wl[2] * (float)g[2] + wl[3] * (float)g[3]
                     + wh[0] * (float)g[4] + wh[1] * (float)g[5]
                     + wh[2] * (float)g[6] + wh[3] * (float)g[7];
            }
            if (lane < 16) Ys[s] -= 0.1f * acc;
            else           Cs[s] -= 0.1f * acc;
        }
        // lanes exchange y/c through LDS across iterations: make the update visible
        asm volatile("s_wait_dscnt 0" ::: "memory");
    }

    if (lane < 16)
        outg[sampleBase + wave * TILE + lane] = Ys[lane];
}

extern "C" void kernel_launch(void* const* d_in, const int* in_sizes, int n_in,
                              void* d_out, int out_size, void* d_ws, size_t ws_size,
                              hipStream_t stream) {
    const float* x  = (const float*)d_in[0];
    const float* W0 = (const float*)d_in[1];
    const float* b0 = (const float*)d_in[2];
    const float* W1 = (const float*)d_in[3];
    const float* b1 = (const float*)d_in[4];
    const float* W2 = (const float*)d_in[5];
    const float* b2 = (const float*)d_in[6];
    const float* W3 = (const float*)d_in[7];
    const float* b3 = (const float*)d_in[8];
    (void)b3; (void)d_ws; (void)ws_size; (void)n_in; (void)out_size;

    const int batch  = in_sizes[0];
    const int blocks = batch / SAMPLES_PER_WG;      // 262144 / 128 = 2048
    ContextEBM_50525995270535_kernel<<<dim3(blocks), dim3(NTHREADS), SMEM_BYTES, stream>>>(
        x, W0, b0, W1, b1, W2, b2, W3, b3, (float*)d_out);
}
